// BahdanauAttention_15994458211112
// MI455X (gfx1250) — matrix operand order repaired and checked by reading, not compiled, not run on hardware
//
#include <hip/hip_runtime.h>
#include <hip/hip_bf16.h>
#include <math.h>

#define B_ 32
#define S_ 2048
#define D_ 1024
#define U_ 512

typedef __attribute__((ext_vector_type(16))) __bf16 v16bf;
typedef __attribute__((ext_vector_type(8)))  float  v8f;

union FragB16 { uint4 q[2]; v16bf v; };

__device__ __forceinline__ unsigned short f2bf(float f) {
    unsigned int u = __float_as_uint(f);
    u += 0x7FFFu + ((u >> 16) & 1u);          // round-to-nearest-even
    return (unsigned short)(u >> 16);
}
__device__ __forceinline__ unsigned int pack2(float lo, float hi) {
    return (unsigned int)f2bf(lo) | ((unsigned int)f2bf(hi) << 16);
}

// ---------------------------------------------------------------------------
// Kernel 0: convert W2 [1024,512] f32 -> bf16 B-fragments in workspace.
// Fragment (nt, kt): 32 lanes x 16 bf16 contiguous per lane.
// Lane L: N = nt*16 + (L&15); K = kt*32 + (L>>4)*16 + j, j=0..15.
// ---------------------------------------------------------------------------
__global__ void __launch_bounds__(256)
prep_w2(const float* __restrict__ W2, unsigned short* __restrict__ w2b) {
    int gid = blockIdx.x * 256 + threadIdx.x;    // 32768 = 32 nt * 32 kt * 32 lanes
    int nt = gid >> 10;
    int kt = (gid >> 5) & 31;
    int L  = gid & 31;
    int k0 = kt * 32 + ((L >> 4) << 4);
    int n  = nt * 16 + (L & 15);
    unsigned short* dst = w2b + (size_t)gid * 16;
#pragma unroll
    for (int j = 0; j < 16; ++j)
        dst[j] = f2bf(W2[(size_t)(k0 + j) * U_ + n]);
}

// ---------------------------------------------------------------------------
// Kernel 1: query[b][u] = decoder_hidden[b] . W1[:,u] + b1[u]   (32 x 512)
// ---------------------------------------------------------------------------
__global__ void __launch_bounds__(512)
prep_query(const float* __restrict__ dec, const float* __restrict__ W1,
           const float* __restrict__ b1, float* __restrict__ q) {
    __shared__ float ds[D_];
    int b = blockIdx.x, t = threadIdx.x;
    ds[t]       = dec[(size_t)b * D_ + t];
    ds[t + 512] = dec[(size_t)b * D_ + t + 512];
    __syncthreads();
    float acc = 0.f;
#pragma unroll 8
    for (int k = 0; k < D_; ++k)
        acc += ds[k] * W1[(size_t)k * U_ + t];
    q[(size_t)b * U_ + t] = acc + b1[t];
}

// ---------------------------------------------------------------------------
// Kernel 2 (main): fused values-GEMM + tanh + score.
// Block = 256 threads (8 waves), handles one (batch b, 32-row s-tile):
// two 16-row M-tiles so every B fragment feeds TWO wmma ops.
// Wave w covers N-tiles w*4 .. w*4+3. kt loop: 32 steps of K=32.
// ---------------------------------------------------------------------------
__global__ void __launch_bounds__(256)
scores_kernel(const float* __restrict__ enc, const unsigned short* __restrict__ w2b,
              const float* __restrict__ q, const float* __restrict__ b2,
              const float* __restrict__ Vv, const float* __restrict__ bv,
              float* __restrict__ scores) {
    // A fragments for both M-tiles: [mt][kt][lane][16] bf16 = 64 KB
    __shared__ __align__(16) unsigned short As[2 * 32 * 32 * 16];
    __shared__ float scW[8][32];

    const int tid   = threadIdx.x;
    const int lane  = tid & 31;
    const int wv    = tid >> 5;
    const int stile = blockIdx.x;   // 0..63  (32 rows each)
    const int b     = blockIdx.y;   // 0..31

    const float* encBase = enc + ((size_t)b * S_ + (size_t)stile * 32) * D_;

    // Stage A tile [32 x 1024] f32 -> bf16 in fragment order.
    // group g (0..4095): mt = g>>11, gg = g&2047, kt = gg>>6, L = (gg>>1)&31, h = gg&1.
#pragma unroll
    for (int i = 0; i < 16; ++i) {
        int g   = tid + i * 256;
        int mt  = g >> 11;
        int gg  = g & 2047;
        int kt  = gg >> 6;
        int L   = (gg >> 1) & 31;
        int h   = gg & 1;
        int row = mt * 16 + (L & 15);
        int col = kt * 32 + ((L >> 4) << 4) + h * 8;
        const float4* src = (const float4*)(encBase + (size_t)row * D_ + col);
        float4 a0 = src[0], a1 = src[1];
        uint4 p;
        p.x = pack2(a0.x, a0.y); p.y = pack2(a0.z, a0.w);
        p.z = pack2(a1.x, a1.y); p.w = pack2(a1.z, a1.w);
        *(uint4*)&As[(size_t)g * 8] = p;     // 16B store
    }
    __syncthreads();

    v8f acc[2][4];
#pragma unroll
    for (int m = 0; m < 2; ++m)
#pragma unroll
        for (int t = 0; t < 4; ++t) acc[m][t] = (v8f){};

    const int ntBase = wv * 4;
    for (int kt = 0; kt < 32; ++kt) {
        FragB16 a0f, a1f;
        const uint4* ap = (const uint4*)&As[((size_t)kt * 32 + lane) * 16];
        a0f.q[0] = ap[0];          a0f.q[1] = ap[1];
        a1f.q[0] = ap[0 + 1024];   a1f.q[1] = ap[1 + 1024];   // +1024 uint4 = +8192 ushorts
#pragma unroll
        for (int t = 0; t < 4; ++t) {
            FragB16 bf;
            const uint4* bp = (const uint4*)(w2b +
                (((size_t)(ntBase + t) * 32 + kt) * 32 + lane) * 16);
            bf.q[0] = bp[0]; bf.q[1] = bp[1];
            acc[0][t] = __builtin_amdgcn_wmma_f32_16x16x32_bf16(
                false, a0f.v, false, bf.v, (short)0, acc[0][t], false, false);
            acc[1][t] = __builtin_amdgcn_wmma_f32_16x16x32_bf16(
                false, a1f.v, false, bf.v, (short)0, acc[1][t], false, false);
        }
    }

    // Epilogue: x = tanh(values + b2 + query); partial score += x * V[n].
    // C layout: VGPR r, lane L -> row M = r + 8*(L>>4), N = L&15.
    float rp[2][8];
#pragma unroll
    for (int m = 0; m < 2; ++m)
#pragma unroll
        for (int r = 0; r < 8; ++r) rp[m][r] = 0.f;

    const int nlane = lane & 15;
#pragma unroll
    for (int t = 0; t < 4; ++t) {
        int n = (ntBase + t) * 16 + nlane;
        float off = b2[n] + q[(size_t)b * U_ + n];
        float vv  = Vv[n];
#pragma unroll
        for (int m = 0; m < 2; ++m)
#pragma unroll
            for (int r = 0; r < 8; ++r)
                rp[m][r] += tanhf(acc[m][t][r] + off) * vv;
    }

    // Reduce across the 16 N-lanes (xor stays within each half-wave).
#pragma unroll
    for (int msk = 1; msk < 16; msk <<= 1) {
#pragma unroll
        for (int m = 0; m < 2; ++m)
#pragma unroll
            for (int r = 0; r < 8; ++r)
                rp[m][r] += __shfl_xor(rp[m][r], msk, 32);
    }
    if (nlane == 0) {
        int half = lane >> 4;                 // rows 0-7 or 8-15 of each M-tile
#pragma unroll
        for (int m = 0; m < 2; ++m)
#pragma unroll
            for (int r = 0; r < 8; ++r)
                scW[wv][m * 16 + half * 8 + r] = rp[m][r];
    }
    __syncthreads();

    if (tid < 32) {
        float s = bv[0];
#pragma unroll
        for (int w = 0; w < 8; ++w) s += scW[w][tid];
        scores[(size_t)b * S_ + (size_t)stile * 32 + tid] = s;
    }
}

// ---------------------------------------------------------------------------
// Kernel 3: in-place softmax over S per batch (scores -> weights in d_out).
// ---------------------------------------------------------------------------
__global__ void __launch_bounds__(256)
softmax_kernel(float* __restrict__ w) {
    __shared__ float red[256];
    int b = blockIdx.x, t = threadIdx.x;
    float* row = w + (size_t)b * S_;
    float v[8];
    float mx = -1e30f;
#pragma unroll
    for (int i = 0; i < 8; ++i) { v[i] = row[t + i * 256]; mx = fmaxf(mx, v[i]); }
    red[t] = mx; __syncthreads();
    for (int o = 128; o > 0; o >>= 1) {
        if (t < o) red[t] = fmaxf(red[t], red[t + o]);
        __syncthreads();
    }
    mx = red[0]; __syncthreads();
    float s = 0.f;
#pragma unroll
    for (int i = 0; i < 8; ++i) { v[i] = expf(v[i] - mx); s += v[i]; }
    red[t] = s; __syncthreads();
    for (int o = 128; o > 0; o >>= 1) {
        if (t < o) red[t] += red[t + o];
        __syncthreads();
    }
    float inv = 1.f / red[0];
#pragma unroll
    for (int i = 0; i < 8; ++i) row[t + i * 256] = v[i] * inv;
}

// ---------------------------------------------------------------------------
// Kernel 4: partial context sums over 128-row s-chunks (deterministic).
// Each thread owns one float4 of the 1024-dim context.
// ---------------------------------------------------------------------------
__global__ void __launch_bounds__(256)
ctx_partial(const float* __restrict__ enc, const float* __restrict__ w,
            float* __restrict__ part) {
    int c = blockIdx.x;   // s-chunk 0..15 (128 rows)
    int b = blockIdx.y;
    int t = threadIdx.x;  // float4 index 0..255
    float4 acc = make_float4(0.f, 0.f, 0.f, 0.f);
    const float* er = enc + ((size_t)b * S_ + (size_t)c * 128) * D_;
    const float* wr = w + (size_t)b * S_ + (size_t)c * 128;
#pragma unroll 4
    for (int s = 0; s < 128; ++s) {
        float ww = wr[s];
        float4 e = ((const float4*)(er + (size_t)s * D_))[t];
        acc.x += ww * e.x; acc.y += ww * e.y;
        acc.z += ww * e.z; acc.w += ww * e.w;
    }
    ((float4*)(part + ((size_t)b * 16 + c) * D_))[t] = acc;
}

// ---------------------------------------------------------------------------
// Kernel 5: reduce the 16 partials -> context in d_out.
// ---------------------------------------------------------------------------
__global__ void __launch_bounds__(256)
ctx_reduce(const float* __restrict__ part, float* __restrict__ ctx) {
    int b = blockIdx.x, t = threadIdx.x;   // t = float4 index 0..255
    float4 s = make_float4(0.f, 0.f, 0.f, 0.f);
#pragma unroll
    for (int c = 0; c < 16; ++c) {
        float4 p = ((const float4*)(part + ((size_t)b * 16 + c) * D_))[t];
        s.x += p.x; s.y += p.y; s.z += p.z; s.w += p.w;
    }
    ((float4*)(ctx + (size_t)b * D_))[t] = s;
}

extern "C" void kernel_launch(void* const* d_in, const int* in_sizes, int n_in,
                              void* d_out, int out_size, void* d_ws, size_t ws_size,
                              hipStream_t stream) {
    const float* dec = (const float*)d_in[0];
    const float* enc = (const float*)d_in[1];
    const float* W1  = (const float*)d_in[2];
    const float* b1  = (const float*)d_in[3];
    const float* W2  = (const float*)d_in[4];
    const float* b2  = (const float*)d_in[5];
    const float* Vv  = (const float*)d_in[6];
    const float* bv  = (const float*)d_in[7];

    char* ws = (char*)d_ws;
    float*          qws  = (float*)ws;                              // 64 KB
    unsigned short* w2b  = (unsigned short*)(ws + 65536);           // 1 MB
    float*          part = (float*)(ws + 65536 + 1048576);          // 2 MB

    float* ctx = (float*)d_out;            // [32, 1024]
    float* wts = (float*)d_out + B_ * D_;  // [32, 2048] (scores -> weights)

    prep_w2      <<<128, 256, 0, stream>>>(W2, w2b);
    prep_query   <<<B_, 512, 0, stream>>>(dec, W1, b1, qws);
    scores_kernel<<<dim3(64, B_), 256, 0, stream>>>(enc, w2b, qws, b2, Vv, bv, wts);
    softmax_kernel<<<B_, 256, 0, stream>>>(wts);
    ctx_partial  <<<dim3(16, B_), 256, 0, stream>>>(enc, wts, part);
    ctx_reduce   <<<B_, 256, 0, stream>>>(part, ctx);
}
